// net_prot_gat_14611478741207
// MI455X (gfx1250) — matrix operand order repaired and checked by reading, not compiled, not run on hardware
//
#include <hip/hip_runtime.h>
#include <cstdint>

// Problem constants (match reference)
#define BB  64
#define NN_ 512
#define DD  256
#define LL  4

typedef __bf16 bf16t;
typedef __attribute__((ext_vector_type(16))) __bf16 v16bf;
typedef __attribute__((ext_vector_type(8)))  __bf16 v8bf;
typedef __attribute__((ext_vector_type(8)))  float  v8f;
typedef __attribute__((ext_vector_type(4)))  int    v4i;

// CDNA5 async global->LDS staging path (bypasses VGPRs, tracked by ASYNCcnt)
#if defined(__gfx1250__) && \
    __has_builtin(__builtin_amdgcn_global_load_async_to_lds_b128) && \
    __has_builtin(__builtin_amdgcn_s_wait_asynccnt)
#define USE_ASYNC_LDS 1
#else
#define USE_ASYNC_LDS 0
#endif

#if USE_ASYNC_LDS
typedef __attribute__((address_space(1))) v4i g_v4i;   // global int4*
typedef __attribute__((address_space(3))) v4i l_v4i;   // LDS int4*
// generic LDS pointer: low 32 bits are the LDS offset (aperture semantics)
static __device__ __forceinline__ l_v4i* to_lds(const void* p) {
    return (l_v4i*)(uint32_t)(uintptr_t)p;
}
static __device__ __forceinline__ g_v4i* to_glb(const void* p) {
    return (g_v4i*)(uintptr_t)p;
}
#endif

// ---------------------------------------------------------------------------
// Tiled bf16 WMMA GEMM:  out[M,Nn] = A[M,K] * B^T  where B is stored [Nn,K]
// row-major ("n-rows contiguous over k"): matches x@W.T, q@x^T, S@xT directly.
// Macro tile 128x128x32, 256 threads = 8 waves, wave tile 32x64 = 2x4 WMMAs.
// Double-buffered LDS; async global->LDS copies overlap next tile with WMMA.
// ---------------------------------------------------------------------------
constexpr int BM = 128, BN = 128, BK = 32, LDL = 40;
constexpr int TILE = BM * LDL;   // elems per matrix tile buffer

// Issue staging for one 128x32 k-chunk of A and B into the given LDS buffers.
static __device__ __forceinline__
void stage_tile(const bf16t* __restrict__ A, int lda, int m0,
                const bf16t* __restrict__ Bm, int ldb, int n0,
                int kk, bf16t* as, bf16t* bs, int t)
{
#pragma unroll
    for (int i = 0; i < 2; ++i) {
        int idx = t + i * 256;          // 0..511
        int row = idx >> 2;             // 0..127
        int c8  = (idx & 3) << 3;       // 0,8,16,24
        const bf16t* ga = A  + (size_t)(m0 + row) * lda + kk + c8;
        const bf16t* gb = Bm + (size_t)(n0 + row) * ldb + kk + c8;
        bf16t* la = as + row * LDL + c8;
        bf16t* lb = bs + row * LDL + c8;
#if USE_ASYNC_LDS
        __builtin_amdgcn_global_load_async_to_lds_b128(to_glb(ga), to_lds(la), 0, 0);
        __builtin_amdgcn_global_load_async_to_lds_b128(to_glb(gb), to_lds(lb), 0, 0);
#else
        *(v8bf*)la = *(const v8bf*)ga;
        *(v8bf*)lb = *(const v8bf*)gb;
#endif
    }
}

// Epilogues:
// 0: +bias -> bf16 | 1: score (sigmoid,*adj,diag) -> f32 | 2: none -> bf16
// 3: +bias,relu -> bf16 | 4: +bias,relu,+resid -> f32 | 5: +bias -> f32
template<int EPI>
__global__ __launch_bounds__(256)
void gemm_bf16(const bf16t* __restrict__ Aall, long sA, int lda,
               const bf16t* __restrict__ Ball, long sB, int ldb,
               int M, int Nn, int K,
               float* __restrict__ outF, long sOutF,
               bf16t* __restrict__ outB, long sOutB,
               int ldc,
               const float* __restrict__ bias,
               const float* __restrict__ resid,
               const float* __restrict__ adjB, long sAdj)
{
    __shared__ __align__(16) bf16t As[2 * TILE];
    __shared__ __align__(16) bf16t Bs[2 * TILE];

    const int z    = blockIdx.z;
    const bf16t* A  = Aall + (size_t)z * sA;
    const bf16t* Bm = Ball + (size_t)z * sB;
    const int m0   = blockIdx.x * BM;
    const int n0   = blockIdx.y * BN;
    const int t    = threadIdx.x;
    const int lane = t & 31;
    const int w    = t >> 5;
    const int wm   = (w & 3) * 32;   // wave row offset in macro tile
    const int wn   = (w >> 2) * 64;  // wave col offset in macro tile
    const int r16  = lane & 15;
    const int half = lane >> 4;

    v8f acc[2][4] = {};

    const int steps = K / BK;
    // prologue: stage tile 0 into buffer 0
    stage_tile(A, lda, m0, Bm, ldb, n0, 0, As, Bs, t);

    for (int s = 0; s < steps; ++s) {
        bf16t* as = As + (s & 1) * TILE;
        bf16t* bs = Bs + (s & 1) * TILE;
#if USE_ASYNC_LDS
        __builtin_amdgcn_s_wait_asynccnt(0);   // our tile-s copies landed
#endif
        __syncthreads();                        // everyone's copies landed
        if (s + 1 < steps)                      // overlap next tile with WMMA
            stage_tile(A, lda, m0, Bm, ldb, n0, (s + 1) * BK,
                       As + ((s + 1) & 1) * TILE, Bs + ((s + 1) & 1) * TILE, t);

        // A fragment (16x32 bf16): lane holds row r16, K = {8h..8h+7, 16+8h..}
        v16bf af[2];
#pragma unroll
        for (int mi = 0; mi < 2; ++mi) {
            const bf16t* p = as + (wm + mi * 16 + r16) * LDL + half * 8;
            v8bf lo = *(const v8bf*)p;
            v8bf hi = *(const v8bf*)(p + 16);
            af[mi] = __builtin_shufflevector(lo, hi,
                       0,1,2,3,4,5,6,7,8,9,10,11,12,13,14,15);
        }
        // B fragment (32x16 bf16): lane holds col r16, K = {8h..8h+7, 16+8h..}
        v16bf bfr[4];
#pragma unroll
        for (int ni = 0; ni < 4; ++ni) {
            const bf16t* p = bs + (wn + ni * 16 + r16) * LDL + half * 8;
            v8bf lo = *(const v8bf*)p;
            v8bf hi = *(const v8bf*)(p + 16);
            bfr[ni] = __builtin_shufflevector(lo, hi,
                       0,1,2,3,4,5,6,7,8,9,10,11,12,13,14,15);
        }
#pragma unroll
        for (int mi = 0; mi < 2; ++mi)
#pragma unroll
            for (int ni = 0; ni < 4; ++ni)
                acc[mi][ni] = __builtin_amdgcn_wmma_f32_16x16x32_bf16(
                    false, af[mi], false, bfr[ni],
                    (short)0, acc[mi][ni], false, false);
        // no trailing barrier: next iteration's top barrier protects reuse
    }

    // Epilogue. C/D layout: lane -> N = lane&15 (+tile), VGPR g -> M = 8*half+g
#pragma unroll
    for (int mi = 0; mi < 2; ++mi) {
#pragma unroll
        for (int ni = 0; ni < 4; ++ni) {
            int nt = n0 + wn + ni * 16 + r16;
            int mb = m0 + wm + mi * 16 + half * 8;
#pragma unroll
            for (int g = 0; g < 8; ++g) {
                int m = mb + g;
                float v = acc[mi][ni][g];
                size_t o = (size_t)m * ldc + nt;
                if constexpr (EPI == 0) {
                    v += bias[nt];
                    outB[(size_t)z * sOutB + o] = (bf16t)v;
                } else if constexpr (EPI == 1) {
                    float s  = 1.0f / (1.0f + __expf(-v));
                    float am = (m == nt) ? 1.0f : adjB[(size_t)z * sAdj + o];
                    s = (s + ((m == nt) ? 1e-5f : 0.0f)) * am;
                    outF[(size_t)z * sOutF + o] = s;
                } else if constexpr (EPI == 2) {
                    outB[(size_t)z * sOutB + o] = (bf16t)v;
                } else if constexpr (EPI == 3) {
                    v += bias[nt];
                    v = v > 0.0f ? v : 0.0f;
                    outB[(size_t)z * sOutB + o] = (bf16t)v;
                } else if constexpr (EPI == 4) {
                    v += bias[nt];
                    v = v > 0.0f ? v : 0.0f;
                    v += resid[o];
                    outF[o] = v;
                } else {
                    v += bias[nt];
                    outF[(size_t)z * sOutF + o] = v;
                }
            }
        }
    }
}

// f32 x[B,N,D] -> bf16 xb[B,N,D] and bf16 xT[B,D,N]
__global__ void cvt_transpose(const float* __restrict__ xin,
                              bf16t* __restrict__ xb,
                              bf16t* __restrict__ xTb)
{
    __shared__ float tile[32][33];
    int b  = blockIdx.z;
    int i0 = blockIdx.x * 32;   // node index
    int d0 = blockIdx.y * 32;   // feature index
    int tx = threadIdx.x, ty = threadIdx.y;   // 32 x 8
    const float* xp = xin + (size_t)b * NN_ * DD;
#pragma unroll
    for (int r = 0; r < 4; ++r) {
        int row = ty + r * 8;
        float v = xp[(size_t)(i0 + row) * DD + d0 + tx];
        tile[row][tx] = v;
        xb[(size_t)b * NN_ * DD + (size_t)(i0 + row) * DD + d0 + tx] = (bf16t)v;
    }
    __syncthreads();
#pragma unroll
    for (int r = 0; r < 4; ++r) {
        int drow = ty + r * 8;
        xTb[(size_t)b * DD * NN_ + (size_t)(d0 + drow) * NN_ + i0 + tx] =
            (bf16t)tile[tx][drow];
    }
}

__global__ void cvt_f32_bf16(const float* __restrict__ s,
                             bf16t* __restrict__ d, int n)
{
    int i = blockIdx.x * 256 + threadIdx.x;
    if (i < n) d[i] = (bf16t)s[i];
}

// Deterministic per-row normalize of raw masked scores (rows of length 512):
// writes normalized f32 back (attls output) and bf16 copy for next GEMM.
__global__ __launch_bounds__(256)
void normalize_rows(float* __restrict__ att, bf16t* __restrict__ Sb)
{
    __shared__ float red[256];
    size_t row = blockIdx.x;            // b*N + i
    float* p = att + row * NN_;
    int t = threadIdx.x;
    float v0 = p[t], v1 = p[t + 256];
    red[t] = v0 + v1;
    __syncthreads();
    for (int off = 128; off > 0; off >>= 1) {
        if (t < off) red[t] += red[t + off];
        __syncthreads();
    }
    float inv = 1.0f / red[0];
    float a0 = v0 * inv, a1 = v1 * inv;
    p[t] = a0; p[t + 256] = a1;
    Sb[row * NN_ + t]       = (bf16t)a0;
    Sb[row * NN_ + t + 256] = (bf16t)a1;
}

extern "C" void kernel_launch(void* const* d_in, const int* in_sizes, int n_in,
                              void* d_out, int out_size, void* d_ws, size_t ws_size,
                              hipStream_t stream)
{
    const float* x     = (const float*)d_in[0];
    const float* adj   = (const float*)d_in[1];
    const float* wattn = (const float*)d_in[2];
    const float* battn = (const float*)d_in[3];
    const float* wl0   = (const float*)d_in[4];
    const float* bl0   = (const float*)d_in[5];
    const float* wl1   = (const float*)d_in[6];
    const float* bl1   = (const float*)d_in[7];
    const float* wfin  = (const float*)d_in[8];
    const float* bfin  = (const float*)d_in[9];

    float* outX    = (float*)d_out;                       // [B,N,D]
    float* attbase = outX + (size_t)BB * NN_ * DD;        // [L,B,N,N]

    const size_t BND = (size_t)BB * NN_ * DD;
    const size_t BNN = (size_t)BB * NN_ * NN_;

    char* wsp = (char*)d_ws;
    auto alloc = [&](size_t bytes) {
        char* p = wsp;
        wsp += (bytes + 255) & ~(size_t)255;
        return p;
    };
    float* xcur = (float*)alloc(BND * 4);      // fp32 activation master
    bf16t* xb   = (bf16t*)alloc(BND * 2);      // bf16 of x
    bf16t* xTb  = (bf16t*)alloc(BND * 2);      // bf16 x^T per batch
    bf16t* qb   = (bf16t*)alloc(BND * 2);      // q, then h1
    bf16t* hb   = (bf16t*)alloc(BND * 2);      // h2
    bf16t* Sb   = (bf16t*)alloc(BNN * 2);      // normalized attention, bf16
    bf16t* wab  = (bf16t*)alloc((size_t)LL * DD * DD * 2);
    bf16t* w0b  = (bf16t*)alloc((size_t)LL * DD * DD * 2);
    bf16t* w1b  = (bf16t*)alloc((size_t)LL * DD * DD * 2);
    bf16t* wfb  = (bf16t*)alloc((size_t)DD * DD * 2);

    (void)hipMemcpyAsync(xcur, x, BND * 4, hipMemcpyDeviceToDevice, stream);

    int nw = LL * DD * DD;
    cvt_f32_bf16<<<(nw + 255) / 256, 256, 0, stream>>>(wattn, wab, nw);
    cvt_f32_bf16<<<(nw + 255) / 256, 256, 0, stream>>>(wl0, w0b, nw);
    cvt_f32_bf16<<<(nw + 255) / 256, 256, 0, stream>>>(wl1, w1b, nw);
    cvt_f32_bf16<<<(DD * DD + 255) / 256, 256, 0, stream>>>(wfin, wfb, DD * DD);

    dim3 blk(256);
    const long NDl = (long)NN_ * DD, NNl = (long)NN_ * NN_, DNl = (long)DD * NN_;

    for (int l = 0; l < LL; ++l) {
        cvt_transpose<<<dim3(16, 8, BB), dim3(32, 8), 0, stream>>>(xcur, xb, xTb);

        // q = x @ Wattn^T + b   (flat M = B*N)
        gemm_bf16<0><<<dim3(BB * NN_ / BM, DD / BN, 1), blk, 0, stream>>>(
            xb, 0, DD, wab + (size_t)l * DD * DD, 0, DD,
            BB * NN_, DD, DD, nullptr, 0, qb, 0, DD,
            battn + (size_t)l * DD, nullptr, nullptr, 0);

        // raw masked scores -> attls_l (f32, unnormalized)
        float* attl = attbase + (size_t)l * BNN;
        gemm_bf16<1><<<dim3(NN_ / BM, NN_ / BN, BB), blk, 0, stream>>>(
            qb, NDl, DD, xb, NDl, DD,
            NN_, NN_, DD, attl, NNl, nullptr, 0, NN_,
            nullptr, nullptr, adj, NNl);

        // row-normalize in place + bf16 copy
        normalize_rows<<<dim3(BB * NN_), 256, 0, stream>>>(attl, Sb);

        // h1 = S @ x  (B-side is xT rows)  -> qb (bf16)
        gemm_bf16<2><<<dim3(NN_ / BM, DD / BN, BB), blk, 0, stream>>>(
            Sb, NNl, NN_, xTb, DNl, NN_,
            NN_, DD, NN_, nullptr, 0, qb, NDl, DD,
            nullptr, nullptr, nullptr, 0);

        // h2 = relu(h1 @ lin0^T + b0) -> hb (bf16), flat M
        gemm_bf16<3><<<dim3(BB * NN_ / BM, DD / BN, 1), blk, 0, stream>>>(
            qb, 0, DD, w0b + (size_t)l * DD * DD, 0, DD,
            BB * NN_, DD, DD, nullptr, 0, hb, 0, DD,
            bl0 + (size_t)l * DD, nullptr, nullptr, 0);

        // xcur = relu(h2 @ lin1^T + b1) + xcur  (f32, in-place)
        gemm_bf16<4><<<dim3(BB * NN_ / BM, DD / BN, 1), blk, 0, stream>>>(
            hb, 0, DD, w1b + (size_t)l * DD * DD, 0, DD,
            BB * NN_, DD, DD, xcur, 0, nullptr, 0, DD,
            bl1 + (size_t)l * DD, xcur, nullptr, 0);
    }

    // final: out = xcur @ final^T + b
    cvt_transpose<<<dim3(16, 8, BB), dim3(32, 8), 0, stream>>>(xcur, xb, xTb);
    gemm_bf16<5><<<dim3(BB * NN_ / BM, DD / BN, 1), blk, 0, stream>>>(
        xb, 0, DD, wfb, 0, DD,
        BB * NN_, DD, DD, outX, 0, nullptr, 0, DD,
        bfin, nullptr, nullptr, 0);
}
